// MultiScale_GraphConv_42485816492179
// MI455X (gfx1250) — matrix operand order repaired and checked by reading, not compiled, not run on hardware
//
#include <hip/hip_runtime.h>
#include <hip/hip_bf16.h>

// ---------------------------------------------------------------------------
// MultiScale GraphConv fused kernel for gfx1250 (MI455X, wave32, WMMA).
//   stage1: support[(s,c),(t,v)] = sum_u (A_powers+A_res)[(s,v),u] * x[n,c,t,u]
//           -> done with v_wmma_f32_16x16x32_bf16, result tile kept in LDS
//   stage2: y[o,(t,v)] = sum_{s,c} W[o,(s,c)] * support   (K = 1248)
//           -> dense WMMA bf16, f32 accumulate, 3 M-tiles x 5 N-tiles / wave
//   epilogue: eval-BN folded into per-o scale/bias, ReLU, coalesced stores
// ---------------------------------------------------------------------------

typedef __attribute__((ext_vector_type(16))) __bf16 v16bf;
typedef __attribute__((ext_vector_type(8)))  float  v8f;

#define V25   25
#define S13   13
#define CIN   96
#define COUT  384
#define NN    16
#define TT    288
#define KTOT  (S13 * CIN)      // 1248
#define COLS  (TT * V25)       // 7200 per batch
#define JB    80               // columns per workgroup (5 wmma N-tiles)
#define NTMAX 5                // t-span of an 80-column block is always 4 (pad 5)
#define SUPK  104              // padded k-stride of support tile (bytes*2 = 208)

__launch_bounds__(256, 2)
__global__ void msgcn_fused_kernel(const float* __restrict__ x,
                                   const float* __restrict__ Ap,
                                   const float* __restrict__ Ar,
                                   const float* __restrict__ W,
                                   const float* __restrict__ bvec,
                                   const float* __restrict__ gamma,
                                   const float* __restrict__ beta,
                                   const float* __restrict__ rmean,
                                   const float* __restrict__ rvar,
                                   float* __restrict__ out) {
  __shared__ __bf16 xbf[CIN * NTMAX * V25];   // 24000 B  x[n] slice, bf16
  __shared__ __bf16 Abf[S13 * V25 * V25];     // 16250 B  A_powers+A_res, bf16
  __shared__ __bf16 sup[JB * SUPK];           // 16640 B  support tile (one s)
  __shared__ float  sscale[COUT];             //  1536 B  BN scale
  __shared__ float  sbias[COUT];              //  1536 B  BN bias  (total < 60KB)

  const int tid  = threadIdx.x;
  const int lane = tid & 31;
  const int wave = tid >> 5;                  // 8 waves
  const int hi   = (lane >= 16) ? 1 : 0;
  const int lm   = lane & 15;

  const int n  = blockIdx.y;                  // batch
  const int j0 = blockIdx.x * JB;             // column block within this batch
  const int t_base = j0 / V25;
  const int nt = (j0 + JB - 1) / V25 - t_base + 1;   // == 4 for JB=80

  // ---------------- prologue staging ----------------
  const float* xg = x + (size_t)n * CIN * TT * V25;
  for (int idx = tid; idx < CIN * nt * V25; idx += 256) {
    int u   = idx % V25;
    int rem = idx / V25;
    int tl  = rem % nt;
    int c   = rem / nt;
    xbf[(c * NTMAX + tl) * V25 + u] =
        (__bf16)xg[((size_t)c * TT + t_base + tl) * V25 + u];
  }
  for (int idx = tid; idx < S13 * V25 * V25; idx += 256) {
    Abf[idx] = (__bf16)(Ap[idx] + Ar[idx]);
  }
  for (int o = tid; o < COUT; o += 256) {
    float sc = gamma[o] * rsqrtf(rvar[o] + 1e-5f);
    sscale[o] = sc;
    sbias[o]  = (bvec[o] - rmean[o]) * sc + beta[o];
  }
  __syncthreads();

  // ---------------- accumulators: 3 M-tiles x 5 N-tiles per wave ----------
  v8f acc[3][5];
#pragma unroll
  for (int m = 0; m < 3; ++m)
#pragma unroll
    for (int q = 0; q < 5; ++q)
#pragma unroll
      for (int i = 0; i < 8; ++i) acc[m][q][i] = 0.0f;

  const __bf16 bz = (__bf16)0.0f;

  for (int s = 0; s < S13; ++s) {
    // prefetch this s's W block for this wave's rows (hint only)
    __builtin_prefetch(W + (size_t)(wave * 48 + lane) * KTOT + s * CIN, 0, 2);

    // ---- stage 1 on matrix cores: P[c,v] = X[c,u] * A_s^T[u,v] ----------
    // units: (c-tile cm in 0..5) x (t-local tl in 0..nt-1), round-robin waves
    for (int unit = wave; unit < 6 * nt; unit += 8) {
      int cm = unit % 6;
      int tl = unit / 6;
      int c  = cm * 16 + lm;
      // A-fragment (16-bit A 16x32 layout): K = e + (e>=8)*8 + hi*8
      v16bf ax;
#pragma unroll
      for (int e = 0; e < 16; ++e) {
        int u = e + ((e >= 8) ? 8 : 0) + hi * 8;
        ax[e] = (u < V25) ? xbf[(c * NTMAX + tl) * V25 + u] : bz;
      }
      int t_g = t_base + tl;
#pragma unroll
      for (int vt = 0; vt < 2; ++vt) {
        int v   = vt * 16 + lm;
        int row = s * V25 + ((v < V25) ? v : (V25 - 1));
        // B-fragment (16-bit B 32x16 layout): K = e + (e>=8)*8 + hi*8, N = lane&15
        v16bf bA;
#pragma unroll
        for (int e = 0; e < 16; ++e) {
          int u = e + ((e >= 8) ? 8 : 0) + hi * 8;
          bA[e] = (u < V25 && v < V25) ? Abf[row * V25 + u] : bz;
        }
        v8f P = {0.f, 0.f, 0.f, 0.f, 0.f, 0.f, 0.f, 0.f};
        P = __builtin_amdgcn_wmma_f32_16x16x32_bf16(
            false, ax, false, bA, (short)0, P, false, false);
        // scatter C-tile to sup[j][k] (C layout: M = i + hi*8, N = lane&15)
#pragma unroll
        for (int i = 0; i < 8; ++i) {
          int k  = cm * 16 + i + hi * 8;         // 0..95 within this s
          int jl = t_g * V25 + v - j0;
          if (v < V25 && jl >= 0 && jl < JB)
            sup[jl * SUPK + k] = (__bf16)P[i];
        }
      }
    }
    __syncthreads();

    // ---- stage 2: dense GEMM over this s's 96 K values (3 steps of 32) ---
#pragma unroll
    for (int kk3 = 0; kk3 < 3; ++kk3) {
      int k0 = s * CIN + kk3 * 32;
      v16bf aw[3];
#pragma unroll
      for (int mm = 0; mm < 3; ++mm) {
        int o = wave * 48 + mm * 16 + lm;
        const float* wp = W + (size_t)o * KTOT + k0 + hi * 16;
        float4 w0 = *(const float4*)(wp);
        float4 w1 = *(const float4*)(wp + 4);
        float4 w2 = *(const float4*)(wp + 8);
        float4 w3 = *(const float4*)(wp + 12);
        aw[mm][0]  = (__bf16)w0.x; aw[mm][1]  = (__bf16)w0.y;
        aw[mm][2]  = (__bf16)w0.z; aw[mm][3]  = (__bf16)w0.w;
        aw[mm][4]  = (__bf16)w1.x; aw[mm][5]  = (__bf16)w1.y;
        aw[mm][6]  = (__bf16)w1.z; aw[mm][7]  = (__bf16)w1.w;
        aw[mm][8]  = (__bf16)w2.x; aw[mm][9]  = (__bf16)w2.y;
        aw[mm][10] = (__bf16)w2.z; aw[mm][11] = (__bf16)w2.w;
        aw[mm][12] = (__bf16)w3.x; aw[mm][13] = (__bf16)w3.y;
        aw[mm][14] = (__bf16)w3.z; aw[mm][15] = (__bf16)w3.w;
      }
#pragma unroll
      for (int q = 0; q < 5; ++q) {
        int jl = q * 16 + lm;
        const __bf16* sp = &sup[jl * SUPK + kk3 * 32 + hi * 16];
        v16bf bs;
#pragma unroll
        for (int e = 0; e < 16; ++e) bs[e] = sp[e];   // 2x ds_load_b128
#pragma unroll
        for (int mm = 0; mm < 3; ++mm)
          acc[mm][q] = __builtin_amdgcn_wmma_f32_16x16x32_bf16(
              false, aw[mm], false, bs, (short)0, acc[mm][q], false, false);
      }
    }
    __syncthreads();   // sup reused next s
  }

  // ---------------- epilogue: BN + ReLU + store ----------------
  float* outn = out + (size_t)n * COUT * COLS;
#pragma unroll
  for (int mm = 0; mm < 3; ++mm) {
    float sc[8], bi[8];
#pragma unroll
    for (int i = 0; i < 8; ++i) {
      int o = wave * 48 + mm * 16 + i + hi * 8;
      sc[i] = sscale[o];
      bi[i] = sbias[o];
    }
#pragma unroll
    for (int q = 0; q < 5; ++q) {
      int jg = j0 + q * 16 + lm;
#pragma unroll
      for (int i = 0; i < 8; ++i) {
        int o = wave * 48 + mm * 16 + i + hi * 8;
        float val = fmaf(acc[mm][q][i], sc[i], bi[i]);
        outn[(size_t)o * COLS + jg] = fmaxf(val, 0.0f);
      }
    }
  }
}

extern "C" void kernel_launch(void* const* d_in, const int* in_sizes, int n_in,
                              void* d_out, int out_size, void* d_ws, size_t ws_size,
                              hipStream_t stream) {
  (void)in_sizes; (void)n_in; (void)d_ws; (void)ws_size; (void)out_size;
  const float* x     = (const float*)d_in[0];
  const float* Ap    = (const float*)d_in[1];
  const float* Ar    = (const float*)d_in[2];
  const float* W     = (const float*)d_in[3];
  const float* b     = (const float*)d_in[4];
  const float* gamma = (const float*)d_in[5];
  const float* beta  = (const float*)d_in[6];
  const float* rmean = (const float*)d_in[7];
  const float* rvar  = (const float*)d_in[8];
  float* out = (float*)d_out;

  dim3 grid(COLS / JB, NN);   // (90, 16)
  dim3 block(256);
  msgcn_fused_kernel<<<grid, block, 0, stream>>>(
      x, Ap, Ar, W, b, gamma, beta, rmean, rvar, out);
}
